// ODConv2D_58712202936585
// MI455X (gfx1250) — compile-verified, matrix operand order checked by reading, not run on hardware
//
#include <hip/hip_runtime.h>
#include <hip/hip_bf16.h>

// ---------------------------------------------------------------------------
// ODConv fused pipeline for MI455X (gfx1250, wave32, WMMA).
// Pointwise 1x1 conv (the 34.4-GFLOP hot spot) runs on v_wmma_f32_16x16x32_f16
// with f32 accumulation; everything else stays fp32.
// ---------------------------------------------------------------------------

typedef __attribute__((ext_vector_type(16))) _Float16 v16h;
typedef __attribute__((ext_vector_type(8)))  _Float16 v8h;
typedef __attribute__((ext_vector_type(8)))  float    v8f;

#define B_    32
#define CIN   256
#define COUT  512
#define HW    64
#define NC    6

__device__ __forceinline__ float sigmoidf_(float v) { return 1.0f / (1.0f + __expf(-v)); }

__device__ __forceinline__ float dot256(const float* __restrict__ w,
                                        const float* __restrict__ c) {
  float s = 0.f;
  #pragma unroll 8
  for (int k = 0; k < 256; ++k) s = fmaf(w[k], c[k], s);
  return s;
}

// ------------------- Kernel 1: context = mean over H*W ---------------------
__global__ void k_context(const float* __restrict__ x, float* __restrict__ ctx) {
  const int bc = blockIdx.x;                       // b*CIN + c  (8192 blocks)
  const float4* p = (const float4*)(x + (size_t)bc * (HW * HW));
  float s = 0.f;
  #pragma unroll
  for (int j = 0; j < 4; ++j) {
    float4 v = p[threadIdx.x + j * 256];
    s += v.x + v.y + v.z + v.w;
  }
  __shared__ float red[256];
  red[threadIdx.x] = s;
  __syncthreads();
  for (int o = 128; o > 0; o >>= 1) {
    if (threadIdx.x < o) red[threadIdx.x] += red[threadIdx.x + o];
    __syncthreads();
  }
  if (threadIdx.x == 0) ctx[bc] = red[0] * (1.0f / (HW * HW));
}

// ------------- Kernel 2: attention heads + fused dynamic weights -----------
__global__ void k_attn(const float* __restrict__ ctx,
                       const float* __restrict__ bank,   // [6,256,1,3,3]
                       const float* __restrict__ w_s, const float* __restrict__ b_s,
                       const float* __restrict__ w_i, const float* __restrict__ b_i,
                       const float* __restrict__ w_k, const float* __restrict__ b_k,
                       const float* __restrict__ w_c, const float* __restrict__ b_c,
                       const float* __restrict__ w_o, const float* __restrict__ b_o,
                       float* __restrict__ dyn,          // [B,256,9]
                       float* __restrict__ ao) {         // [B,512]
  const int b = blockIdx.x, t = threadIdx.x;
  __shared__ float s_ctx[CIN];
  __shared__ float s_ai[CIN];
  __shared__ float s_s9[9];
  __shared__ float s_cand[NC];
  s_ctx[t] = ctx[b * CIN + t];
  __syncthreads();

  // attn_in (sigmoid), per channel
  s_ai[t] = sigmoidf_(dot256(w_i + (size_t)t * CIN, s_ctx) + b_i[t]);
  // attn_out (sigmoid), 2 rows per thread -> global scratch
  ao[b * COUT + t]       = sigmoidf_(dot256(w_o + (size_t)t * CIN, s_ctx) + b_o[t]);
  ao[b * COUT + t + 256] = sigmoidf_(dot256(w_o + (size_t)(t + 256) * CIN, s_ctx) + b_o[t + 256]);
  if (t < 9)  s_s9[t]   = sigmoidf_(dot256(w_s + (size_t)t * CIN, s_ctx) + b_s[t]);
  if (t < NC) s_cand[t] = dot256(w_c + (size_t)t * CIN, s_ctx) + b_c[t];
  __syncthreads();

  if (t == 0) {  // softmax over 6 candidates
    float mx = s_cand[0];
    for (int n = 1; n < NC; ++n) mx = fmaxf(mx, s_cand[n]);
    float sum = 0.f;
    for (int n = 0; n < NC; ++n) { s_cand[n] = __expf(s_cand[n] - mx); sum += s_cand[n]; }
    const float inv = 1.0f / sum;
    for (int n = 0; n < NC; ++n) s_cand[n] *= inv;
  }
  __syncthreads();

  // dyn[c][kk] = (sum_n cand[n]*bank[n][c][kk]) * s9[kk] * ai[c] * ak[c][kk]
  const int c = t;
  #pragma unroll
  for (int kk = 0; kk < 9; ++kk) {
    const float akv = sigmoidf_(dot256(w_k + (size_t)(c * 9 + kk) * CIN, s_ctx) + b_k[c * 9 + kk]);
    float dd = 0.f;
    #pragma unroll
    for (int n = 0; n < NC; ++n)
      dd = fmaf(s_cand[n], bank[((size_t)n * CIN + c) * 9 + kk], dd);
    dyn[((size_t)b * CIN + c) * 9 + kk] = dd * s_s9[kk] * s_ai[c] * akv;
  }
}

// ------------- Kernel 3: pw_w fp32 -> f16 (lives in L2, 256 KB) ------------
__global__ void k_cvt_f16(const float* __restrict__ w, _Float16* __restrict__ o) {
  const int i = blockIdx.x * 256 + threadIdx.x;    // 512 blocks -> 131072 elems
  o[i] = (_Float16)w[i];
}

// ---- Kernel 4: fused depthwise conv + WMMA pointwise GEMM + output scale ---
// One workgroup per (b, h): produces out[b, :, h, :]  (512 x 64).
__global__ void __launch_bounds__(256)
k_main(const float* __restrict__ x,
       const float* __restrict__ dyn,
       const _Float16* __restrict__ pwh,   // [512,256] f16
       const float* __restrict__ pw_b,
       const float* __restrict__ ao,
       float* __restrict__ out) {
  const int h = blockIdx.x;                // 0..63
  const int b = blockIdx.y;                // 0..31
  const int tid = threadIdx.x;

  // dw tile, pixel-major so K (channel) is contiguous -> b128 LDS fragment loads
  __shared__ _Float16 lds_dw[HW * CIN];    // 32 KB

  // ---- Phase 1: depthwise 3x3 conv (fp32), one channel per thread --------
  {
    const int c = tid;
    float wgt[9];
    #pragma unroll
    for (int j = 0; j < 9; ++j) wgt[j] = dyn[((size_t)b * CIN + c) * 9 + j];

    float acc[HW];
    #pragma unroll
    for (int p = 0; p < HW; ++p) acc[p] = 0.f;

    const float* xc = x + (((size_t)b * CIN + c) * HW + h) * HW;  // row h
    #pragma unroll
    for (int dy = -1; dy <= 1; ++dy) {
      const int hh = h + dy;
      if (hh < 0 || hh > HW - 1) continue;           // zero-pad rows
      const float* r = xc + dy * HW;
      const float w0 = wgt[(dy + 1) * 3 + 0];
      const float w1 = wgt[(dy + 1) * 3 + 1];
      const float w2 = wgt[(dy + 1) * 3 + 2];
      float xm = 0.f, xc0 = r[0];                    // rolling window, zero-pad cols
      #pragma unroll
      for (int p = 0; p < HW; ++p) {
        const float xp = (p < HW - 1) ? r[p + 1] : 0.f;
        acc[p] = fmaf(xm, w0, fmaf(xc0, w1, fmaf(xp, w2, acc[p])));
        xm = xc0; xc0 = xp;
      }
    }
    #pragma unroll
    for (int p = 0; p < HW; ++p) lds_dw[p * CIN + c] = (_Float16)acc[p];
  }
  __syncthreads();

  // ---- Phase 2: out[o,p] = sum_c pw[o,c]*dw[c,p], via WMMA f16 -> f32 ----
  // ISA 16-bit A layout: lane(0..15)=row M, lane group hi=lane>>4 picks K+8;
  // VGPR 0-3 <-> K[base..base+8), VGPR 4-7 <-> K[base+16..base+24).
  const int wave = tid >> 5;
  const int lane = tid & 31;
  const int mrow = lane & 15;   // A row / B col / D col within the 16-tile
  const int hi   = lane >> 4;   // K sub-slab selector

  for (int mt = wave; mt < COUT / 16; mt += 8) {     // 4 M-tiles per wave
    v8f acc[4] = {v8f{}, v8f{}, v8f{}, v8f{}};
    const _Float16* pA = pwh + (size_t)(mt * 16 + mrow) * CIN;

    #pragma unroll
    for (int kt = 0; kt < CIN / 32; ++kt) {          // K loop: 8 x 32 channels
      const int k0 = kt * 32 + hi * 8;
      const v8h a0 = *(const v8h*)(pA + k0);
      const v8h a1 = *(const v8h*)(pA + k0 + 16);
      v16h a;
      #pragma unroll
      for (int i = 0; i < 8; ++i) { a[i] = a0[i]; a[i + 8] = a1[i]; }

      #pragma unroll
      for (int nt = 0; nt < 4; ++nt) {               // 4 N-tiles (64 pixels)
        const int p = nt * 16 + mrow;
        const v8h b0 = *(const v8h*)&lds_dw[p * CIN + k0];
        const v8h b1 = *(const v8h*)&lds_dw[p * CIN + k0 + 16];
        v16h bm;
        #pragma unroll
        for (int i = 0; i < 8; ++i) { bm[i] = b0[i]; bm[i + 8] = b1[i]; }
        acc[nt] = __builtin_amdgcn_wmma_f32_16x16x32_f16(
            /*neg_a=*/false, a, /*neg_b=*/false, bm,
            /*c_mod=*/(short)0, acc[nt], /*reuse_a=*/false, /*reuse_b=*/false);
      }
    }

    // Epilogue: (acc + pw_b[o]) * attn_out[b,o].
    // C/D layout: VGPR r holds M = r + 8*hi, N = mrow.
    float bias[8], scale[8];
    #pragma unroll
    for (int r = 0; r < 8; ++r) {
      const int o = mt * 16 + hi * 8 + r;
      bias[r]  = pw_b[o];
      scale[r] = ao[b * COUT + o];
    }
    #pragma unroll
    for (int nt = 0; nt < 4; ++nt) {
      const int p = nt * 16 + mrow;
      #pragma unroll
      for (int r = 0; r < 8; ++r) {
        const int o = mt * 16 + hi * 8 + r;
        out[(((size_t)b * COUT + o) * HW + h) * HW + p] = (acc[nt][r] + bias[r]) * scale[r];
      }
    }
  }
}

// ---------------------------------------------------------------------------
extern "C" void kernel_launch(void* const* d_in, const int* in_sizes, int n_in,
                              void* d_out, int out_size, void* d_ws, size_t ws_size,
                              hipStream_t stream) {
  const float* x    = (const float*)d_in[0];
  const float* bank = (const float*)d_in[1];
  const float* pw_w = (const float*)d_in[2];
  const float* pw_b = (const float*)d_in[3];
  const float* w_s  = (const float*)d_in[4];
  const float* b_s  = (const float*)d_in[5];
  const float* w_i  = (const float*)d_in[6];
  const float* b_i  = (const float*)d_in[7];
  const float* w_k  = (const float*)d_in[8];
  const float* b_k  = (const float*)d_in[9];
  const float* w_c  = (const float*)d_in[10];
  const float* b_c  = (const float*)d_in[11];
  const float* w_o  = (const float*)d_in[12];
  const float* b_o  = (const float*)d_in[13];
  float* out = (float*)d_out;

  // Scratch layout (bytes):
  //   [0,       32768)  ctx   [32,256]  f32
  //   [32768,   98304)  ao    [32,512]  f32
  //   [98304,  393216)  dyn   [32,256,9] f32
  //   [393216, 655360)  pw_w  [512,256] f16
  char* wsb = (char*)d_ws;
  float*    g_ctx = (float*)(wsb + 0);
  float*    g_ao  = (float*)(wsb + 32768);
  float*    g_dyn = (float*)(wsb + 98304);
  _Float16* g_pwh = (_Float16*)(wsb + 393216);

  k_context<<<dim3(B_ * CIN), dim3(256), 0, stream>>>(x, g_ctx);
  k_attn<<<dim3(B_), dim3(256), 0, stream>>>(g_ctx, bank,
                                             w_s, b_s, w_i, b_i, w_k, b_k,
                                             w_c, b_c, w_o, b_o,
                                             g_dyn, g_ao);
  k_cvt_f16<<<dim3((COUT * CIN) / 256), dim3(256), 0, stream>>>(pw_w, g_pwh);
  k_main<<<dim3(HW, B_), dim3(256), 0, stream>>>(x, g_dyn, g_pwh, pw_b, g_ao, out);
}
